// Expansion_7791070675483
// MI455X (gfx1250) — matrix operand order repaired and checked by reading, not compiled, not run on hardware
//
#include <hip/hip_runtime.h>
#include <hip/hip_bf16.h>

// ---------------------------------------------------------------------------
// Expansion (e3nn tensor-product expansion) for MI455X / gfx1250.
//
// Reformulated as one bf16 WMMA GEMM:
//   C[4096, 4352] = G_sel[4096, 1056] @ L[1056, 4352]
// G[b, k*16+w] = h'[b,k] * x[b,w]   (h' = silu MLP hidden, + const-1 slot
//                                    absorbing lb2, + one zero pad slot)
// L[k*16+w, n] = lw2'[k, flat_weight_index(w, n)]  (pure column permutation)
// Epilogue scatters C into the 80x80 output blocks with e3nn scales/biases.
// ---------------------------------------------------------------------------

typedef __attribute__((ext_vector_type(16))) __bf16 v16bf;
typedef __attribute__((ext_vector_type(8)))  float  v8f;

struct __align__(16) U4 { unsigned int x, y, z, w; };

#define NB    4096   // batch
#define KDIM  1056   // 66 k-groups * 16 w  (64 hidden + 1 bias + 1 zero pad)
#define NTOT  4352   // 1024 (blk00) + 256 (blk11) + 3*512 (blk01) + 3*512 (blk10)

__device__ __forceinline__ float silu_f(float x) { return x / (1.0f + __expf(-x)); }

// ---------------- prep: hidden layers of both MLPs -------------------------
__global__ void k_hidden(const float* __restrict__ emb,
                         const float* __restrict__ lw1, const float* __restrict__ lb1,
                         const float* __restrict__ bw1, const float* __restrict__ bb1,
                         float* __restrict__ Hf, float* __restrict__ Hb) {
    __shared__ float e[128];
    const int b = blockIdx.x, t = threadIdx.x;
    e[t] = emb[b * 128 + t];
    __syncthreads();
    if (t < 64) {
        float acc = lb1[t];
        #pragma unroll 8
        for (int k = 0; k < 128; ++k) acc += e[k] * lw1[k * 64 + t];
        Hf[b * 66 + t] = silu_f(acc);
        if (t == 0) { Hf[b * 66 + 64] = 1.0f; Hf[b * 66 + 65] = 0.0f; }
    } else {
        const int u = t - 64;
        float acc = bb1[u];
        #pragma unroll 8
        for (int k = 0; k < 128; ++k) acc += e[k] * bw1[k * 64 + u];
        Hb[b * 64 + u] = silu_f(acc);
    }
}

// ---------------- prep: per-sample irreps x0, x1 ---------------------------
__global__ void k_x(const float* __restrict__ feat,
                    const float* __restrict__ W0, const float* __restrict__ W1,
                    float* __restrict__ x0, float* __restrict__ x1) {
    const int b = blockIdx.x, t = threadIdx.x;
    const float* fr = feat + (size_t)b * 320;
    if (t < 16) {
        float acc = 0.f;
        #pragma unroll 8
        for (int u = 0; u < 128; ++u) acc += fr[u] * W0[u * 16 + t];
        x0[b * 16 + t] = acc * 0.08838834764831845f;          // 1/sqrt(128)
    } else {
        const int q = t - 16, vo = q / 3, j = q % 3;          // 48 outputs
        float acc = 0.f;
        #pragma unroll 8
        for (int w = 0; w < 64; ++w) acc += fr[128 + w * 3 + j] * W1[w * 16 + vo];
        x1[b * 48 + vo * 3 + j] = acc * 0.125f;               // 1/sqrt(64)
    }
}

// ---------------- prep: bias vector bb = silu(..)@bw2 + bb2 ----------------
__global__ void k_bias(const float* __restrict__ Hb,
                       const float* __restrict__ bw2, const float* __restrict__ bb2,
                       float* __restrict__ biasv) {
    const int tid = blockIdx.x * blockDim.x + threadIdx.x;
    const int n = tid % 1280, b = tid / 1280;
    const float* h = Hb + b * 64;
    float acc = bb2[n];
    #pragma unroll 8
    for (int k = 0; k < 64; ++k) acc += h[k] * bw2[k * 1280 + n];
    biasv[(size_t)b * 1280 + n] = acc;
}

// ---------------- prep: A matrices G[4][NB][KDIM] (bf16) -------------------
__global__ void k_G(const float* __restrict__ Hf,
                    const float* __restrict__ x0, const float* __restrict__ x1,
                    __bf16* __restrict__ G) {
    const size_t tid = (size_t)blockIdx.x * blockDim.x + threadIdx.x;
    const int K = (int)(tid % KDIM);
    const size_t rest = tid / KDIM;
    const int b = (int)(rest % NB);
    const int g = (int)(rest / NB);                 // 0: x0 path, 1..3: x1[:, :, g-1]
    const int k = K >> 4, w = K & 15;
    const float h = Hf[b * 66 + k];
    const float x = (g == 0) ? x0[b * 16 + w] : x1[b * 48 + w * 3 + (g - 1)];
    G[tid] = (__bf16)(h * x);
}

// ---------------- prep: B matrix L[NTOT][KDIM] (bf16, N-major) -------------
__global__ void k_B(const float* __restrict__ lw2, const float* __restrict__ lb2,
                    __bf16* __restrict__ Bm) {
    const size_t tid = (size_t)blockIdx.x * blockDim.x + threadIdx.x;
    const int K = (int)(tid % KDIM);
    const int n = (int)(tid / KDIM);
    const int k = K >> 4, w = K & 15;
    int col;
    if (n < 1024)       col = w * 1024 + n;                         // blk00 (U*32+V)
    else if (n < 1280)  col = 16384 + w * 256 + (n - 1024);         // blk11 (u*16+v)
    else if (n < 2816)  col = 20480 + w * 512 + ((n - 1280) & 511); // blk01 (u*16+v), j=(n-1280)>>9
    else                col = 28672 + w * 512 + ((n - 2816) & 511); // blk10 (u*32+v), i=(n-2816)>>9
    float v;
    if (k < 64)       v = lw2[(size_t)k * 36864 + col];
    else if (k == 64) v = lb2[col];   // constant-1 slot of h' carries lb2
    else              v = 0.0f;       // zero pad slot
    Bm[tid] = (__bf16)v;
}

// ---------------- main: bf16 WMMA GEMM + scatter epilogue ------------------
__global__ __launch_bounds__(256) void k_gemm(
        const __bf16* __restrict__ G, const __bf16* __restrict__ Bm,
        const float* __restrict__ biasv, float* __restrict__ out) {
    // 96-byte padded rows: 16B-aligned b128 LDS accesses, staggered banks.
    __shared__ __align__(16) __bf16 Asm[128][48];
    __shared__ __align__(16) __bf16 Bsm[128][48];

    const int n0 = blockIdx.x * 128;
    const int m0 = blockIdx.y * 128;
    const int t = threadIdx.x;
    const int wave = t >> 5, lane = t & 31;
    const int mi = wave & 3, ni = wave >> 2;     // 4x2 wave grid over 128x128 tile
    const int g = lane >> 4, lm = lane & 15;

    // A-matrix select (uniform per block: all region boundaries are 128-aligned)
    const int sel = (n0 < 1280) ? 0 : (1 + (((n0 - 1280) >> 9) % 3));
    const __bf16* Ag = G + (size_t)sel * NB * KDIM;

    // cooperative staging: each thread moves one 32B segment of A and B
    const int row = t >> 1, seg = t & 1;
    const __bf16* gA = Ag + (size_t)(m0 + row) * KDIM + seg * 16;
    const __bf16* gB = Bm + (size_t)(n0 + row) * KDIM + seg * 16;

    v8f acc[2][4];
    #pragma unroll
    for (int a = 0; a < 2; ++a)
        #pragma unroll
        for (int c = 0; c < 4; ++c)
            #pragma unroll
            for (int e = 0; e < 8; ++e) acc[a][c][e] = 0.0f;

    for (int kk = 0; kk < KDIM; kk += 32) {
        __syncthreads();
        *(U4*)&Asm[row][seg * 16] = *(const U4*)(gA + kk);
        *(U4*)&Asm[row][seg * 16 + 8] = *(const U4*)(gA + kk + 8);
        *(U4*)&Bsm[row][seg * 16] = *(const U4*)(gB + kk);
        *(U4*)&Bsm[row][seg * 16 + 8] = *(const U4*)(gB + kk + 8);
        if (kk + 32 < KDIM) {               // pull next K-slab toward L2/L0
            __builtin_prefetch(gA + kk + 32, 0, 1);
            __builtin_prefetch(gB + kk + 32, 0, 1);
        }
        __syncthreads();

        // Fragment layouts per CDNA5 ISA 7.12.2 (16-bit, wave32):
        //   A (16xK32): lane -> M=lane%16; elem e -> K=(e&7)+16*(e>>3)+8*(lane>>4)
        //   B (K32x16): lane -> N=lane%16; elem e -> K=(e&7)+16*(e>>3)+8*(lane>>4)
        v16bf af[2], bfr[4];
        #pragma unroll
        for (int fm = 0; fm < 2; ++fm) {
            const __bf16* ar = &Asm[mi * 32 + fm * 16 + lm][0];
            U4* ap = (U4*)&af[fm];
            ap[0] = *(const U4*)(ar + 8 * g);        // K = 8g .. 8g+7
            ap[1] = *(const U4*)(ar + 16 + 8 * g);   // K = 16+8g .. 23+8g
        }
        #pragma unroll
        for (int fn = 0; fn < 4; ++fn) {
            const __bf16* br = &Bsm[ni * 64 + fn * 16 + lm][0];
            U4* bp = (U4*)&bfr[fn];
            bp[0] = *(const U4*)(br + 8 * g);        // K = 8g .. 8g+7
            bp[1] = *(const U4*)(br + 16 + 8 * g);   // K = 16+8g .. 23+8g
        }
        #pragma unroll
        for (int fm = 0; fm < 2; ++fm)
            #pragma unroll
            for (int fn = 0; fn < 4; ++fn)
                acc[fm][fn] = __builtin_amdgcn_wmma_f32_16x16x32_bf16(
                    false, af[fm], false, bfr[fn], (short)0, acc[fm][fn],
                    false, false);
    }

    // ---------------- epilogue: scatter into 80x80 blocks ------------------
    const float S00 = 0.0625f;                    // 1/16 (mul normalization)
    const float S3  = 0.036084391824351615f;      // 1/(16*sqrt(3))
    const int region = (n0 < 1024) ? 0 : (n0 < 1280 ? 1 : (n0 < 2816 ? 2 : 3));

    #pragma unroll
    for (int fm = 0; fm < 2; ++fm) {
        #pragma unroll
        for (int fn = 0; fn < 4; ++fn) {
            const int n = n0 + ni * 64 + fn * 16 + lm;
            #pragma unroll
            for (int r = 0; r < 8; ++r) {
                const int b = m0 + mi * 32 + fm * 16 + r + 8 * g;  // C layout: M=r+8*(lane/16)
                const float c = acc[fm][fn][r];
                float* ob = out + (size_t)b * 6400;
                if (region == 0) {                       // blk00 -> [0:32, 0:32]
                    const int U = n >> 5, V = n & 31;
                    ob[U * 80 + V] = (c + biasv[(size_t)b * 1280 + n]) * S00;
                } else if (region == 1) {                // blk11 diag -> [32:80, 32:80]
                    const int mI = n - 1024, u = mI >> 4, v = mI & 15;
                    const float val = (c + biasv[(size_t)b * 1280 + 1024 + mI]) * S3;
                    float* p = ob + (32 + 3 * u) * 80 + (32 + 3 * v);
                    p[0]   = val; p[1]   = 0.f; p[2]   = 0.f;
                    p[80]  = 0.f; p[81]  = val; p[82]  = 0.f;
                    p[160] = 0.f; p[161] = 0.f; p[162] = val;
                } else if (region == 2) {                // blk01 -> [0:32, 32:80]
                    const int r2 = n - 1280, j = r2 >> 9, q = r2 & 511;
                    const int u = q >> 4, v = q & 15;
                    ob[u * 80 + 32 + 3 * v + j] = c * S3;
                } else {                                 // blk10 -> [32:80, 0:32]
                    const int r3 = n - 2816, i = r3 >> 9, q = r3 & 511;
                    const int u = q >> 5, v = q & 31;
                    ob[(32 + 3 * u + i) * 80 + v] = c * S3;
                }
            }
        }
    }
}

// ---------------------------------------------------------------------------
extern "C" void kernel_launch(void* const* d_in, const int* in_sizes, int n_in,
                              void* d_out, int out_size, void* d_ws, size_t ws_size,
                              hipStream_t stream) {
    (void)in_sizes; (void)n_in; (void)out_size; (void)ws_size;
    const float* feat = (const float*)d_in[0];
    const float* emb  = (const float*)d_in[1];
    const float* W0   = (const float*)d_in[2];
    const float* W1   = (const float*)d_in[3];
    const float* lw1  = (const float*)d_in[4];
    const float* lb1  = (const float*)d_in[5];
    const float* lw2  = (const float*)d_in[6];
    const float* lb2  = (const float*)d_in[7];
    const float* bw1  = (const float*)d_in[8];
    const float* bb1  = (const float*)d_in[9];
    const float* bw2  = (const float*)d_in[10];
    const float* bb2  = (const float*)d_in[11];
    float* out = (float*)d_out;

    char* ws = (char*)d_ws;
    float*  Hf    = (float*)ws;   ws += (size_t)NB * 66 * 4;
    float*  Hb    = (float*)ws;   ws += (size_t)NB * 64 * 4;
    float*  x0    = (float*)ws;   ws += (size_t)NB * 16 * 4;
    float*  x1    = (float*)ws;   ws += (size_t)NB * 48 * 4;
    float*  biasv = (float*)ws;   ws += (size_t)NB * 1280 * 4;
    __bf16* G     = (__bf16*)ws;  ws += (size_t)4 * NB * KDIM * 2;
    __bf16* Bm    = (__bf16*)ws;

    k_hidden<<<NB, 128, 0, stream>>>(emb, lw1, lb1, bw1, bb1, Hf, Hb);
    k_x     <<<NB, 64, 0, stream>>>(feat, W0, W1, x0, x1);
    k_bias  <<<(NB * 1280) / 256, 256, 0, stream>>>(Hb, bw2, bb2, biasv);
    k_G     <<<(4 * NB * KDIM) / 256, 256, 0, stream>>>(Hf, x0, x1, G);
    k_B     <<<(NTOT * KDIM) / 256, 256, 0, stream>>>(lw2, lb2, Bm);

    dim3 grid(NTOT / 128, NB / 128);   // 34 x 32 workgroups
    k_gemm<<<grid, 256, 0, stream>>>(G, Bm, biasv, out);
}
